// VectorQuantizer_42271068127602
// MI455X (gfx1250) — matrix operand order repaired and checked by reading, not compiled, not run on hardware
//
#include <hip/hip_runtime.h>
#include <hip/hip_bf16.h>
#include <stdint.h>

// ---------------------------------------------------------------------------
// VQ-VAE vector quantizer for gfx1250 (MI455X), wave32 + WMMA bf16.
// dist argmin via  score = ||c||^2 - 2*(z . c)   (||z||^2 is row-constant).
// z.c computed with bf16x3 split (hi*hi + lo*hi + hi*lo) on
// V_WMMA_F32_16X16X32_BF16 for ~fp32 accuracy at bf16 matrix throughput.
// Codebook chunks double-buffered in LDS via GLOBAL_LOAD_ASYNC_TO_LDS_B128
// (ASYNCcnt), overlapping L2->LDS staging with WMMA compute.
// ---------------------------------------------------------------------------

typedef __attribute__((ext_vector_type(16))) __bf16 v16bf;
typedef __attribute__((ext_vector_type(8)))  float  v8f;

#define N_TOTAL 131072
#define K_CODES 1024
#define DIM     64
#define M_TILE  128          // rows per workgroup (8 waves x 16 rows)
#define KC      128          // codebook codes staged in LDS per chunk
#define BLOCK   256
#define NUM_WG  (N_TOTAL / M_TILE)   // 1024
#define BETA_W  0.25f
#define CHUNK_BYTES (KC * DIM * 2)   // bytes per bf16 plane chunk (16 KB)

// d_ws layout (bytes)
#define WS_CSQ  0                                  // float[1024]
#define WS_HI   4096                               // bf16[1024*64] hi plane
#define WS_LO   (4096 + 131072)                    // bf16[1024*64] lo plane
#define WS_PART (4096 + 131072 + 131072)           // float[NUM_WG] loss partials

__device__ __forceinline__ uint16_t f32_to_bf16_rn(float f) {
  uint32_t u = __float_as_uint(f);
  uint32_t r = u + 0x7FFFu + ((u >> 16) & 1u);     // round-to-nearest-even
  return (uint16_t)(r >> 16);
}
__device__ __forceinline__ float bf16_to_f32(uint16_t h) {
  return __uint_as_float(((uint32_t)h) << 16);
}

// --------------------------------------------------------------------------
// Kernel 1: split codebook into bf16 hi/lo planes + per-code squared norms.
// grid = K_CODES blocks of DIM(=64) threads (2 waves).
// --------------------------------------------------------------------------
__global__ void vq_prep_kernel(const float* __restrict__ cb,
                               float* __restrict__ csq,
                               uint16_t* __restrict__ hiP,
                               uint16_t* __restrict__ loP) {
  const int code = blockIdx.x;
  const int t    = threadIdx.x;
  float f = cb[code * DIM + t];
  uint16_t h = f32_to_bf16_rn(f);
  uint16_t l = f32_to_bf16_rn(f - bf16_to_f32(h));
  hiP[code * DIM + t] = h;
  loP[code * DIM + t] = l;

  float s = f * f;
  #pragma unroll
  for (int off = 1; off < 32; off <<= 1) s += __shfl_xor(s, off, 32);
  __shared__ float wsum[2];
  if ((t & 31) == 0) wsum[t >> 5] = s;
  __syncthreads();
  if (t == 0) csq[code] = wsum[0] + wsum[1];
}

// --------------------------------------------------------------------------
// Kernel 2: main WMMA argmin + gather + per-workgroup loss partial.
// --------------------------------------------------------------------------
__global__ __launch_bounds__(BLOCK)
void vq_main_kernel(const float* __restrict__ z,
                    const float* __restrict__ cb,
                    const float* __restrict__ csq,
                    const uint16_t* __restrict__ hiP,
                    const uint16_t* __restrict__ loP,
                    float* __restrict__ outIdx,
                    float* __restrict__ outZq,
                    float* __restrict__ partial) {
  // [buffer][plane(hi=0,lo=1)][code*DIM + dim]  -> 64 KB ping-pong
  __shared__ __align__(16) uint16_t ldsBuf[2][2][KC * DIM];
  __shared__ float ldsCsq[K_CODES];            // all 1024 norms, loaded once
  __shared__ int   ldsIdx[M_TILE];
  __shared__ float ldsLoss[BLOCK / 32];

  const int tid  = threadIdx.x;
  const int wave = tid >> 5;
  const int lane = tid & 31;
  const int l15  = lane & 15;
  const int sel  = lane >> 4;                 // lane group (0: low half, 1: high half)
  const int rowBase = blockIdx.x * M_TILE;
  const int waveRow = rowBase + wave * 16;
  const int rowA    = waveRow + l15;

  // ---- all code norms to LDS (once)
  #pragma unroll
  for (int j = 0; j < K_CODES / BLOCK; ++j)
    ldsCsq[tid + j * BLOCK] = csq[tid + j * BLOCK];

  union Frag { v16bf v; uint32_t d[8]; uint4 q[2]; };

  // ---- Build A fragments (16x32 bf16, contiguous K to match B):
  // lane<16: row=l15, dwords0-7 hold K=32h..32h+15
  // lane>=16 same row, K shifted by +16. Two K-halves h=0,1 cover d=64.
  Frag aHi[2], aLo[2];
  const float* zr = z + (size_t)rowA * DIM;
  #pragma unroll
  for (int h = 0; h < 2; ++h) {
    #pragma unroll
    for (int c2 = 0; c2 < 2; ++c2) {
      const int base = 32 * h + 16 * sel + c2 * 8;
      float4 f0 = *(const float4*)(zr + base);
      float4 f1 = *(const float4*)(zr + base + 4);
      float fs[8] = {f0.x, f0.y, f0.z, f0.w, f1.x, f1.y, f1.z, f1.w};
      #pragma unroll
      for (int j = 0; j < 4; ++j) {
        uint16_t h0 = f32_to_bf16_rn(fs[2 * j]);
        uint16_t h1 = f32_to_bf16_rn(fs[2 * j + 1]);
        uint16_t q0 = f32_to_bf16_rn(fs[2 * j]     - bf16_to_f32(h0));
        uint16_t q1 = f32_to_bf16_rn(fs[2 * j + 1] - bf16_to_f32(h1));
        aHi[h].d[c2 * 4 + j] = (uint32_t)h0 | ((uint32_t)h1 << 16);
        aLo[h].d[c2 * 4 + j] = (uint32_t)q0 | ((uint32_t)q1 << 16);
      }
    }
  }

  float bestVal[8];
  int   bestIdx[8];
  #pragma unroll
  for (int i = 0; i < 8; ++i) { bestVal[i] = 3.4e38f; bestIdx[i] = 0; }

  // ---- async staging machinery: memory -> LDS, tracked by ASYNCcnt
  const uint32_t ldsHiA[2] = { (uint32_t)(uintptr_t)&ldsBuf[0][0][0],
                               (uint32_t)(uintptr_t)&ldsBuf[1][0][0] };
  const uint32_t ldsLoA[2] = { (uint32_t)(uintptr_t)&ldsBuf[0][1][0],
                               (uint32_t)(uintptr_t)&ldsBuf[1][1][0] };
  const char* hiByte = (const char*)hiP;
  const char* loByte = (const char*)loP;

  auto stage_async = [&](int chunk, int b) {
    const uint64_t sHi = (uint64_t)(uintptr_t)(hiByte + (size_t)chunk * CHUNK_BYTES);
    const uint64_t sLo = (uint64_t)(uintptr_t)(loByte + (size_t)chunk * CHUNK_BYTES);
    #pragma unroll
    for (int j = 0; j < 4; ++j) {
      const uint32_t voff = (uint32_t)(tid * 16 + j * (BLOCK * 16)); // byte offset
      const uint32_t dHi  = ldsHiA[b] + voff;
      const uint32_t dLo  = ldsLoA[b] + voff;
      // GVS mode: 64-bit SGPR base + 32-bit per-lane VGPR offset; vdst = LDS addr
      asm volatile("global_load_async_to_lds_b128 %0, %1, %2"
                   :: "v"(dHi), "v"(voff), "s"(sHi) : "memory");
      asm volatile("global_load_async_to_lds_b128 %0, %1, %2"
                   :: "v"(dLo), "v"(voff), "s"(sLo) : "memory");
    }
  };

  stage_async(0, 0);
  asm volatile("s_wait_asynccnt 0x0" ::: "memory");
  __syncthreads();

  for (int chunk = 0; chunk < K_CODES / KC; ++chunk) {
    const int b = chunk & 1;
    if (chunk + 1 < K_CODES / KC) stage_async(chunk + 1, b ^ 1);  // overlap

    const uint16_t* tileHi = &ldsBuf[b][0][0];
    const uint16_t* tileLo = &ldsBuf[b][1][0];

    #pragma unroll 2
    for (int t = 0; t < KC / 16; ++t) {
      const int cb16 = t * 16;
      // ---- B fragments (32x16 bf16 layout): lane n holds code (cb16 + n%16),
      // lanes 0-15 K=32h..32h+15, lanes 16-31 K=32h+16..32h+31; pairs per dword.
      // LDS tile is [code][dim] row-major bf16 -> 32 contiguous bytes per lane.
      const int boff = (cb16 + l15) * DIM + sel * 16;   // ushort index, 32B aligned
      Frag bH0, bH1, bL0, bL1;
      const uint4* ph0 = (const uint4*)(tileHi + boff);
      const uint4* pl0 = (const uint4*)(tileLo + boff);
      const uint4* ph1 = (const uint4*)(tileHi + boff + 32);
      const uint4* pl1 = (const uint4*)(tileLo + boff + 32);
      bH0.q[0] = ph0[0]; bH0.q[1] = ph0[1];
      bL0.q[0] = pl0[0]; bL0.q[1] = pl0[1];
      bH1.q[0] = ph1[0]; bH1.q[1] = ph1[1];
      bL1.q[0] = pl1[0]; bL1.q[1] = pl1[1];

      v8f acc = {0.f, 0.f, 0.f, 0.f, 0.f, 0.f, 0.f, 0.f};
      // bf16x3: hi*hi + lo*hi + hi*lo  (~fp32-accurate dot product)
      acc = __builtin_amdgcn_wmma_f32_16x16x32_bf16(false, aHi[0].v, false, bH0.v, (short)0, acc, false, false);
      acc = __builtin_amdgcn_wmma_f32_16x16x32_bf16(false, aHi[1].v, false, bH1.v, (short)0, acc, false, false);
      acc = __builtin_amdgcn_wmma_f32_16x16x32_bf16(false, aLo[0].v, false, bH0.v, (short)0, acc, false, false);
      acc = __builtin_amdgcn_wmma_f32_16x16x32_bf16(false, aLo[1].v, false, bH1.v, (short)0, acc, false, false);
      acc = __builtin_amdgcn_wmma_f32_16x16x32_bf16(false, aHi[0].v, false, bL0.v, (short)0, acc, false, false);
      acc = __builtin_amdgcn_wmma_f32_16x16x32_bf16(false, aHi[1].v, false, bL1.v, (short)0, acc, false, false);

      // C/D layout: VGPR i -> row (i + 8*sel), col = l15
      const int   code = chunk * KC + cb16 + l15;
      const float cs   = ldsCsq[chunk * KC + cb16 + l15];
      #pragma unroll
      for (int i = 0; i < 8; ++i) {
        float s = cs - 2.0f * acc[i];
        if (s < bestVal[i]) { bestVal[i] = s; bestIdx[i] = code; }
      }
    }

    // next chunk's async staging must have landed before the barrier releases it
    if (chunk + 1 < K_CODES / KC)
      asm volatile("s_wait_asynccnt 0x0" ::: "memory");
    __syncthreads();
  }

  // ---- argmin reduction across the 16 lanes of each column group
  #pragma unroll
  for (int i = 0; i < 8; ++i) {
    float v = bestVal[i];
    int   ix = bestIdx[i];
    #pragma unroll
    for (int off = 1; off < 16; off <<= 1) {
      float ov = __shfl_xor(v, off, 32);
      int   oi = __shfl_xor(ix, off, 32);
      if (ov < v || (ov == v && oi < ix)) { v = ov; ix = oi; }
    }
    bestIdx[i] = ix;
  }
  if (l15 == 0) {
    #pragma unroll
    for (int i = 0; i < 8; ++i) {
      const int localRow = wave * 16 + sel * 8 + i;
      ldsIdx[localRow] = bestIdx[i];
      outIdx[rowBase + localRow] = (float)bestIdx[i];   // indices as float
    }
  }
  __syncthreads();

  // ---- gather z_q = codebook[idx], accumulate (e - z)^2 partial
  float lsum = 0.f;
  #pragma unroll
  for (int j = 0; j < 8; ++j) {
    const int id  = tid + j * BLOCK;   // 2048 float4 slots = 128 rows x 16
    const int row = id >> 4;
    const int qq  = id & 15;
    const int idx = ldsIdx[row];
    float4 c  = *(const float4*)(cb + (size_t)idx * DIM + qq * 4);
    float4 zv = *(const float4*)(z + (size_t)(rowBase + row) * DIM + qq * 4);
    *(float4*)(outZq + (size_t)(rowBase + row) * DIM + qq * 4) = c;
    float dx = c.x - zv.x, dy = c.y - zv.y, dz = c.z - zv.z, dw = c.w - zv.w;
    lsum += dx * dx + dy * dy + dz * dz + dw * dw;
  }
  #pragma unroll
  for (int off = 1; off < 32; off <<= 1) lsum += __shfl_xor(lsum, off, 32);
  if (lane == 0) ldsLoss[wave] = lsum;
  __syncthreads();
  if (tid == 0) {
    float s = 0.f;
    #pragma unroll
    for (int w = 0; w < BLOCK / 32; ++w) s += ldsLoss[w];
    partial[blockIdx.x] = s;    // deterministic: no float atomics
  }
}

// --------------------------------------------------------------------------
// Kernel 3: fixed-order reduction of loss partials -> scalar vq_loss.
// --------------------------------------------------------------------------
__global__ void vq_loss_reduce(const float* __restrict__ partial,
                               float* __restrict__ outLoss) {
  const int t = threadIdx.x;
  float s = 0.f;
  #pragma unroll
  for (int j = 0; j < NUM_WG / BLOCK; ++j) s += partial[t + j * BLOCK];
  #pragma unroll
  for (int off = 1; off < 32; off <<= 1) s += __shfl_xor(s, off, 32);
  __shared__ float wsum[BLOCK / 32];
  if ((t & 31) == 0) wsum[t >> 5] = s;
  __syncthreads();
  if (t == 0) {
    float tot = 0.f;
    #pragma unroll
    for (int w = 0; w < BLOCK / 32; ++w) tot += wsum[w];
    // codebook_loss and commitment_loss are numerically identical values.
    *outLoss = (1.0f + BETA_W) * tot / (float)(N_TOTAL * DIM);
  }
}

// --------------------------------------------------------------------------
extern "C" void kernel_launch(void* const* d_in, const int* in_sizes, int n_in,
                              void* d_out, int out_size, void* d_ws, size_t ws_size,
                              hipStream_t stream) {
  const float* z  = (const float*)d_in[0];   // (131072, 64) f32
  const float* cb = (const float*)d_in[1];   // (1024, 64)  f32

  float* out     = (float*)d_out;
  float* outIdx  = out;                                   // N indices (as float)
  float* outZq   = out + N_TOTAL;                         // N*64 z_q_st
  float* outLoss = out + N_TOTAL + (size_t)N_TOTAL * DIM; // 1 scalar

  char* ws = (char*)d_ws;
  float*    csq     = (float*)(ws + WS_CSQ);
  uint16_t* hiP     = (uint16_t*)(ws + WS_HI);
  uint16_t* loP     = (uint16_t*)(ws + WS_LO);
  float*    partial = (float*)(ws + WS_PART);

  vq_prep_kernel<<<K_CODES, DIM, 0, stream>>>(cb, csq, hiP, loP);
  vq_main_kernel<<<NUM_WG, BLOCK, 0, stream>>>(z, cb, csq, hiP, loP,
                                               outIdx, outZq, partial);
  vq_loss_reduce<<<1, BLOCK, 0, stream>>>(partial, outLoss);
}
